// PPOLSTM_84078279786813
// MI455X (gfx1250) — matrix operand order repaired and checked by reading, not compiled, not run on hardware
//
#include <hip/hip_runtime.h>
#include <hip/hip_bf16.h>
#include <stdint.h>

// Problem sizes (fixed by the reference)
#define TT   256
#define BB   1024
#define OBSD 64
#define HH   128
#define AA   32
#define G4   512          // 4*H
#define NN   (TT * BB)    // 262144

typedef __attribute__((ext_vector_type(16))) __bf16 bf16x16;
typedef __attribute__((ext_vector_type(8)))  __bf16 bf16x8;
typedef __attribute__((ext_vector_type(8)))  float  f32x8;
typedef __attribute__((ext_vector_type(4)))  float  f32x4;
typedef __bf16 bf16;

static __device__ __forceinline__ int lane_id() { return (int)(threadIdx.x & 31); }

static __device__ __forceinline__ f32x8 splat8(float v) {
  f32x8 c;
#pragma unroll
  for (int i = 0; i < 8; ++i) c[i] = v;
  return c;
}
static __device__ __forceinline__ bf16x16 zero16bf() {
  bf16x16 z;
#pragma unroll
  for (int i = 0; i < 16; ++i) z[i] = (bf16)0.0f;
  return z;
}

// ---------------- CDNA5 WMMA fragment loaders (wave32, CDNA5 ISA §7.12.2) ---------------
// A 16x32 bf16: lane l, row m=l&15; lanes 0-15 hold K={k0..k0+7, k0+16..k0+23},
//               lanes 16-31 hold K={k0+8..k0+15, k0+24..k0+31}.
// Each lane = two contiguous 16B runs -> two ds_load_b128.
static __device__ __forceinline__ bf16x16 load_a_lds(const bf16* base, int ldk, int k0) {
  int l = lane_id();
  const bf16* p = base + (l & 15) * ldk + k0 + ((l >> 4) * 8);
  bf16x8 lo = *(const bf16x8*)p;
  bf16x8 hi = *(const bf16x8*)(p + 16);
  bf16x16 a;
#pragma unroll
  for (int i = 0; i < 8; ++i) { a[i] = lo[i]; a[8 + i] = hi[i]; }
  return a;
}
// A from global fp32 row-major: two 32B runs -> four global_load_b128 + pack to bf16.
static __device__ __forceinline__ bf16x16 load_a_g32(const float* base, int ldk, int k0) {
  int l = lane_id();
  const float* p = base + (long)(l & 15) * ldk + k0 + ((l >> 4) * 8);
  f32x4 a0 = *(const f32x4*)p;
  f32x4 a1 = *(const f32x4*)(p + 4);
  f32x4 b0 = *(const f32x4*)(p + 16);
  f32x4 b1 = *(const f32x4*)(p + 20);
  bf16x16 a;
#pragma unroll
  for (int i = 0; i < 4; ++i) {
    a[i]      = (bf16)a0[i];
    a[4 + i]  = (bf16)a1[i];
    a[8 + i]  = (bf16)b0[i];
    a[12 + i] = (bf16)b1[i];
  }
  return a;
}
// B 32x16 bf16: lane l holds column n=l&15; same K order as A (k0+(l>>4)*8+{0..7, 16..23}).
// Source stored [n][k] row-major -> two contiguous 16B runs -> two ds_load_b128.
static __device__ __forceinline__ bf16x16 load_b_nk(const bf16* base, int ldk, int k0, int n0) {
  int l = lane_id();
  const bf16* p = base + (n0 + (l & 15)) * ldk + k0 + ((l >> 4) * 8);
  bf16x8 lo = *(const bf16x8*)p;
  bf16x8 hi = *(const bf16x8*)(p + 16);
  bf16x16 b;
#pragma unroll
  for (int i = 0; i < 8; ++i) { b[i] = lo[i]; b[8 + i] = hi[i]; }
  return b;
}
// C/D 16x16 f32: lane l col n=l&15, VGPR i = row i + (l>=16 ? 8 : 0).

static __device__ __forceinline__ f32x8 wmma_bf16(bf16x16 a, bf16x16 b, f32x8 c) {
  return __builtin_amdgcn_wmma_f32_16x16x32_bf16(false, a, false, b, (short)0, c, false, false);
}

static __device__ __forceinline__ float fsigmoid(float x) {
  return 1.0f / (1.0f + __expf(-x));
}

// Stage fp32 global -> bf16 LDS, 8 elements/thread/iter (b128 load + b128 store).
static __device__ __forceinline__ void stage_bf16(bf16* dst, const float* src, int count,
                                                  int tid, int nthreads) {
  for (int i = tid; i < count / 8; i += nthreads) {
    f32x4 lo = *(const f32x4*)(src + i * 8);
    f32x4 hi = *(const f32x4*)(src + i * 8 + 4);
    bf16x8 v;
#pragma unroll
    for (int j = 0; j < 4; ++j) { v[j] = (bf16)lo[j]; v[4 + j] = (bf16)hi[j]; }
    ((bf16x8*)dst)[i] = v;
  }
}
// Stage fp32 [k][n] global -> bf16 [n][k] LDS (transpose), contiguous 8-wide k-runs at dst.
static __device__ __forceinline__ void stage_bf16_T(bf16* dst, const float* src,
                                                    int K, int Nc, int tid, int nthreads) {
  const int kchunks = K / 8;
  for (int i = tid; i < Nc * kchunks; i += nthreads) {
    const int nn = i / kchunks;
    const int k8 = (i % kchunks) * 8;
    bf16x8 v;
#pragma unroll
    for (int j = 0; j < 8; ++j) v[j] = (bf16)src[(k8 + j) * Nc + nn];
    *(bf16x8*)(dst + nn * K + k8) = v;
  }
}

// =====================================================================================
// Kernel 1: gates_x = relu(relu(x@W1+b1)@W2+b2) @ w_ih^T + (b_ih+b_hh)     [N, 512]
// 8 waves / block, each wave owns one 16-row tile; weights in LDS as bf16, all in
// [n][k] order so every B fragment is a pair of ds_load_b128.
// =====================================================================================
__global__ void __launch_bounds__(256)
ppolstm_trunk_gates(const float* __restrict__ x,
                    const float* __restrict__ W1, const float* __restrict__ b1,
                    const float* __restrict__ W2, const float* __restrict__ b2,
                    const float* __restrict__ w_ih, const float* __restrict__ b_ih,
                    const float* __restrict__ b_hh,
                    float* __restrict__ gates_x) {
  extern __shared__ char smem_raw[];
  bf16*  sW1t  = (bf16*)smem_raw;            // [128][64]   (W1 transposed)
  bf16*  sW2t  = sW1t + HH * OBSD;           // [128][128]  (W2 transposed)
  bf16*  sWih  = sW2t + HH * HH;             // [512][128]  (w_ih natural [gate][k])
  bf16*  sStg  = sWih + G4 * HH;             // 8 waves * 16*128 staging
  float* sB1   = (float*)(sStg + 8 * 16 * HH);
  float* sB2   = sB1 + HH;
  float* sBg   = sB2 + HH;                   // 512: b_ih + b_hh

  const int tid = threadIdx.x;
  stage_bf16_T(sW1t, W1, OBSD, HH, tid, 256);
  stage_bf16_T(sW2t, W2, HH, HH, tid, 256);
  stage_bf16(sWih, w_ih, G4 * HH, tid, 256);
  for (int i = tid; i < HH; i += 256) { sB1[i] = b1[i]; sB2[i] = b2[i]; }
  for (int i = tid; i < G4; i += 256) sBg[i] = b_ih[i] + b_hh[i];
  __syncthreads();

  const int wave = tid >> 5;
  const int l = lane_id();
  const int n = l & 15;
  const int moff = (l >> 4) * 8;
  const long row0 = (long)blockIdx.x * 128 + wave * 16;
  bf16* stg = sStg + wave * 16 * HH;

  // ---- GEMM1: h1 = relu(x @ W1 + b1), K=64 ----
  bf16x16 xa0 = load_a_g32(x + row0 * OBSD, OBSD, 0);
  bf16x16 xa1 = load_a_g32(x + row0 * OBSD, OBSD, 32);
#pragma unroll
  for (int nt = 0; nt < 8; ++nt) {
    const int n0 = nt * 16;
    f32x8 c = splat8(sB1[n0 + n]);
    c = wmma_bf16(xa0, load_b_nk(sW1t, OBSD, 0,  n0), c);
    c = wmma_bf16(xa1, load_b_nk(sW1t, OBSD, 32, n0), c);
#pragma unroll
    for (int i = 0; i < 8; ++i) stg[(moff + i) * HH + n0 + n] = (bf16)fmaxf(c[i], 0.0f);
  }
  // wave-private staging: same-wave LDS store->load is in order (DScnt).

  // ---- GEMM2: h2 = relu(h1 @ W2 + b2), K=128 ----
  // A fragments hoisted to registers first, so each column tile can be stored to LDS
  // immediately (overwriting h1 is safe).
  bf16x16 ha[4];
#pragma unroll
  for (int kf = 0; kf < 4; ++kf) ha[kf] = load_a_lds(stg, HH, kf * 32);
#pragma unroll
  for (int nt = 0; nt < 8; ++nt) {
    const int n0 = nt * 16;
    f32x8 c = splat8(sB2[n0 + n]);
#pragma unroll
    for (int kf = 0; kf < 4; ++kf)
      c = wmma_bf16(ha[kf], load_b_nk(sW2t, HH, kf * 32, n0), c);
#pragma unroll
    for (int i = 0; i < 8; ++i)
      stg[(moff + i) * HH + n0 + n] = (bf16)fmaxf(c[i], 0.0f);
  }

  // ---- GEMM3: gates = h2 @ w_ih^T + (b_ih + b_hh), 512 output columns ----
#pragma unroll
  for (int kf = 0; kf < 4; ++kf) ha[kf] = load_a_lds(stg, HH, kf * 32);
  for (int gt = 0; gt < 32; ++gt) {
    const int g0 = gt * 16;
    f32x8 c = splat8(sBg[g0 + n]);
#pragma unroll
    for (int kf = 0; kf < 4; ++kf)
      c = wmma_bf16(ha[kf], load_b_nk(sWih, HH, kf * 32, g0), c);
#pragma unroll
    for (int i = 0; i < 8; ++i)
      gates_x[(row0 + moff + i) * G4 + g0 + n] = c[i];
  }
}

// =====================================================================================
// Kernel 2: sequential LSTM scan. 64 persistent blocks x 16 batch rows; rows are
// independent through the recurrence, so no inter-block sync. Wave w owns hidden
// columns [16w, 16w+16); c stays in registers; h ping-pongs through LDS (bf16) so the
// serial loop needs exactly ONE s_barrier per timestep. All 16 w_hh B-fragments
// (128 VGPRs) are hoisted out of the time loop: per step the critical chain is just
// 8 ds_load_b128 -> 16 v_wmma -> pointwise -> 8 ds_store_b16 -> s_barrier.
// =====================================================================================
__global__ void __launch_bounds__(256)
ppolstm_scan(const float* __restrict__ gates_x,
             const int* __restrict__ done,
             const float* __restrict__ h0, const float* __restrict__ c0,
             const float* __restrict__ w_hh,
             float* __restrict__ hid,
             float* __restrict__ hN, float* __restrict__ cN) {
  extern __shared__ char smem_raw[];
  bf16*  sWhh   = (bf16*)smem_raw;               // [512][128]  ([gate][k])
  bf16*  sHbuf  = sWhh + G4 * HH;                // 2 x 16*128 (ping-pong)
  float* sKeepb = (float*)(sHbuf + 2 * 16 * HH); // 2 x 16 (ping-pong)

  const int tid = threadIdx.x;
  const int r0  = blockIdx.x * 16;
  stage_bf16(sWhh, w_hh, G4 * HH, tid, 256);
  stage_bf16(sHbuf, h0 + (long)r0 * HH, 16 * HH, tid, 256);   // buffer 0 = h at t=0
  if (tid < 16) sKeepb[tid] = done[r0 + tid] ? 0.0f : 1.0f;   // keep for t=0

  const int wave = tid >> 5;
  const int l = lane_id();
  const int n = l & 15;
  const int moff = (l >> 4) * 8;
  const int hc0 = wave * 16;                     // owned hidden columns

  f32x8 cst, hlast;
#pragma unroll
  for (int i = 0; i < 8; ++i) {
    cst[i] = c0[(long)(r0 + moff + i) * HH + hc0 + n];
    hlast[i] = 0.0f;
  }
  __syncthreads();

  // Loop-invariant recurrent weights: this wave's 4 gate-tiles x 4 K-slices.
  bf16x16 bfr[4][4];
#pragma unroll
  for (int g = 0; g < 4; ++g)
#pragma unroll
    for (int kf = 0; kf < 4; ++kf)
      bfr[g][kf] = load_b_nk(sWhh, HH, kf * 32, g * HH + hc0);

  for (int t = 0; t < TT; ++t) {
    const long tb = (long)t * BB + r0;
    bf16*  sHc = sHbuf + (t & 1) * 16 * HH;
    bf16*  sHn = sHbuf + ((t + 1) & 1) * 16 * HH;
    const float* keep = sKeepb + (t & 1) * 16;

    // Read full h (all 128 K) into A fragments; apply done-reset (keep in {0,1}).
    bf16x16 hfr[4];
#pragma unroll
    for (int kf = 0; kf < 4; ++kf) hfr[kf] = load_a_lds(sHc, HH, kf * 32);
    if (keep[l & 15] == 0.0f) {
      bf16x16 z = zero16bf();
#pragma unroll
      for (int kf = 0; kf < 4; ++kf) hfr[kf] = z;
    }
#pragma unroll
    for (int i = 0; i < 8; ++i) cst[i] *= keep[moff + i];

    // Gate accumulators start from the precomputed x-part.
    f32x8 acc[4];
#pragma unroll
    for (int g = 0; g < 4; ++g) {
      const float* gp = gates_x + tb * G4 + g * HH + hc0 + n;
#pragma unroll
      for (int i = 0; i < 8; ++i) acc[g][i] = gp[(long)(moff + i) * G4];
    }
    if (t + 1 < TT) {
      const float* pf = gates_x + ((long)(t + 1) * BB + r0 + moff) * G4 + hc0 + n;
#pragma unroll
      for (int g = 0; g < 4; ++g) __builtin_prefetch(pf + g * HH, 0, 0);
    }

    // Recurrent GEMM: [16x128] x [128 x 4*16] via bf16 WMMA, fp32 accumulate.
#pragma unroll
    for (int g = 0; g < 4; ++g)
#pragma unroll
      for (int kf = 0; kf < 4; ++kf)
        acc[g] = wmma_bf16(hfr[kf], bfr[g][kf], acc[g]);

    // Pointwise LSTM cell (torch gate order i,f,g,o) for owned 16x16 hidden tile.
#pragma unroll
    for (int i = 0; i < 8; ++i) {
      const float ig = fsigmoid(acc[0][i]);
      const float fg = fsigmoid(acc[1][i]);
      const float gg = tanhf(acc[2][i]);
      const float og = fsigmoid(acc[3][i]);
      const float cv = fg * cst[i] + ig * gg;
      cst[i] = cv;
      const float hv = og * tanhf(cv);
      hlast[i] = hv;
      sHn[(moff + i) * HH + hc0 + n] = (bf16)hv;                // next step's input
      hid[(tb + moff + i) * HH + hc0 + n] = hv;                 // scan output
    }
    if (t + 1 < TT && tid < 16)
      sKeepb[((t + 1) & 1) * 16 + tid] = done[(long)(t + 1) * BB + r0 + tid] ? 0.0f : 1.0f;

    __syncthreads();   // single barrier: publishes sHn + next keep, retires sHc reads
  }

#pragma unroll
  for (int i = 0; i < 8; ++i) {
    hN[(long)(r0 + moff + i) * HH + hc0 + n] = hlast[i];
    cN[(long)(r0 + moff + i) * HH + hc0 + n] = cst[i];
  }
}

// =====================================================================================
// Kernel 3: heads. logits = hid@Wa + ba (WMMA), masked log-softmax -> lp/ent,
// value = hid@Wc + bc. 8 waves/block, 16 rows per wave. Wa staged transposed.
// =====================================================================================
__global__ void __launch_bounds__(256)
ppolstm_heads(const float* __restrict__ hid,
              const int* __restrict__ action,
              const uint8_t* __restrict__ mask,
              const float* __restrict__ Wa, const float* __restrict__ ba,
              const float* __restrict__ Wc, const float* __restrict__ bc,
              float* __restrict__ lp, float* __restrict__ ent, float* __restrict__ val) {
  __shared__ __align__(16) bf16  sWat[AA * HH];     // [32][128] (Wa transposed)
  __shared__ __align__(16) float sba[AA];
  __shared__ __align__(16) float sWc[HH];
  __shared__ __align__(16) float sLog[8][16 * AA];
  __shared__ float sbc;

  const int tid = threadIdx.x;
  stage_bf16_T(sWat, Wa, HH, AA, tid, 256);
  for (int i = tid; i < HH; i += 256) sWc[i] = Wc[i];
  if (tid < AA) sba[tid] = ba[tid];
  if (tid == 0) sbc = bc[0];
  __syncthreads();

  const int wave = tid >> 5;
  const int l = lane_id();
  const int n = l & 15;
  const int moff = (l >> 4) * 8;
  const long row0 = (long)blockIdx.x * 128 + wave * 16;

  bf16x16 ha[4];
#pragma unroll
  for (int kf = 0; kf < 4; ++kf) ha[kf] = load_a_g32(hid + row0 * HH, HH, kf * 32);

#pragma unroll
  for (int at = 0; at < 2; ++at) {
    const int n0 = at * 16;
    f32x8 c = splat8(sba[n0 + n]);
#pragma unroll
    for (int kf = 0; kf < 4; ++kf)
      c = wmma_bf16(ha[kf], load_b_nk(sWat, HH, kf * 32, n0), c);
#pragma unroll
    for (int i = 0; i < 8; ++i) {
      const long r = row0 + moff + i;
      const float lg = mask[r * AA + n0 + n] ? c[i] : -1e8f;
      sLog[wave][(moff + i) * AA + n0 + n] = lg;
    }
  }
  // Same-wave LDS store->load is ordered; lanes 0-15 each reduce one row.
  if (l < 16) {
    const long gr = row0 + l;
    const float* L = &sLog[wave][l * AA];
    float m = L[0];
#pragma unroll
    for (int j = 1; j < AA; ++j) m = fmaxf(m, L[j]);
    float s = 0.0f;
#pragma unroll
    for (int j = 0; j < AA; ++j) s += __expf(L[j] - m);
    const float logZ = m + __logf(s);
    lp[gr] = L[action[gr]] - logZ;
    float e = 0.0f;
#pragma unroll
    for (int j = 0; j < AA; ++j) {
      const float p = __expf(L[j] - logZ);
      e += p * (logZ - L[j]);
    }
    ent[gr] = e;
    float v = sbc;
    const float* hr = hid + gr * HH;
#pragma unroll
    for (int k = 0; k < HH; k += 4) {
      f32x4 hv = *(const f32x4*)(hr + k);
      f32x4 wv = *(const f32x4*)(sWc + k);
      v += hv[0] * wv[0] + hv[1] * wv[1] + hv[2] * wv[2] + hv[3] * wv[3];
    }
    val[gr] = v;
  }
}

// =====================================================================================
extern "C" void kernel_launch(void* const* d_in, const int* in_sizes, int n_in,
                              void* d_out, int out_size, void* d_ws, size_t ws_size,
                              hipStream_t stream) {
  (void)in_sizes; (void)n_in; (void)out_size; (void)ws_size;
  const float*   x      = (const float*)d_in[0];
  const int*     done   = (const int*)d_in[1];
  const int*     action = (const int*)d_in[2];
  const uint8_t* mask   = (const uint8_t*)d_in[3];
  const float*   h0     = (const float*)d_in[4];
  const float*   c0     = (const float*)d_in[5];
  const float*   W1     = (const float*)d_in[6];
  const float*   b1     = (const float*)d_in[7];
  const float*   W2     = (const float*)d_in[8];
  const float*   b2     = (const float*)d_in[9];
  const float*   w_ih   = (const float*)d_in[10];
  const float*   w_hh   = (const float*)d_in[11];
  const float*   b_ih   = (const float*)d_in[12];
  const float*   b_hh   = (const float*)d_in[13];
  const float*   Wa     = (const float*)d_in[14];
  const float*   ba     = (const float*)d_in[15];
  const float*   Wc     = (const float*)d_in[16];
  const float*   bc     = (const float*)d_in[17];

  float* gates = (float*)d_ws;                       // [N, 512]
  float* hid   = gates + (size_t)NN * G4;            // [N, 128]

  float* out = (float*)d_out;
  float* lp  = out;                                  // [N]
  float* ent = out + (size_t)NN;                     // [N]
  float* val = out + 2 * (size_t)NN;                 // [N, 1]
  float* hN  = out + 3 * (size_t)NN;                 // [1, B, H]
  float* cN  = hN + (size_t)BB * HH;                 // [1, B, H]

  // Dynamic LDS (bf16 weight tiles exceed the 64 KB static limit; WGP has 320 KB).
  const size_t lds1 = (size_t)(HH * OBSD + HH * HH + G4 * HH + 8 * 16 * HH) * sizeof(bf16)
                    + (size_t)(HH + HH + G4) * sizeof(float);
  const size_t lds2 = (size_t)(G4 * HH + 2 * 16 * HH) * sizeof(bf16) + 32 * sizeof(float);

  ppolstm_trunk_gates<<<NN / 128, 256, lds1, stream>>>(x, W1, b1, W2, b2,
                                                       w_ih, b_ih, b_hh, gates);
  ppolstm_scan<<<BB / 16, 256, lds2, stream>>>(gates, done, h0, c0, w_hh, hid, hN, cN);
  ppolstm_heads<<<NN / 128, 256, 0, stream>>>(hid, action, mask, Wa, ba, Wc, bc,
                                              lp, ent, val);
}
